// EMAMixing_41764261986783
// MI455X (gfx1250) — compile-verified, matrix operand order checked by reading, not run on hardware
//
#include <hip/hip_runtime.h>

// Problem sizes (fixed by the reference)
#define L_SEQ 2048
#define BATCH 8
#define DIM   2048                 // D = K = N
#define MROWS (L_SEQ * BATCH)      // 16384 GEMM rows

typedef __bf16 bf16_t;
typedef bf16_t v16bf __attribute__((ext_vector_type(16)));
typedef bf16_t v8bf  __attribute__((ext_vector_type(8)));
typedef bf16_t v4bf  __attribute__((ext_vector_type(4)));
typedef float  v8f   __attribute__((ext_vector_type(8)));
typedef float  v4f   __attribute__((ext_vector_type(4)));
typedef int    v4i_t __attribute__((ext_vector_type(4)));

// Async global->LDS path (CDNA5 GLOBAL_LOAD_ASYNC_TO_LDS_B128, ASYNCcnt).
// Guarded: device pass only, and only if the builtin exists on this toolchain.
// Probe-derived signature: (v4i addrspace(1)* src, v4i addrspace(3)* dst, imm, imm)
#if defined(__gfx1250__) && __has_builtin(__builtin_amdgcn_global_load_async_to_lds_b128)
#define USE_ASYNC_LDS 1
#else
#define USE_ASYNC_LDS 0
#endif

#if USE_ASYNC_LDS
typedef __attribute__((address_space(1))) v4i_t* gas_ptr;
typedef __attribute__((address_space(3))) v4i_t* las_ptr;
#endif

// ---------------------------------------------------------------------------
// Kernel 1: EMA scan. conv[t] = f*conv[t-1] + x[t];
// x_mix[t] = x_mix_last*f^(t+1) + (1-f)*conv[t], stored bf16 as GEMM A[M,K].
// One thread per (b,d): loads/stores coalesced across d (stride B*D per step).
// ---------------------------------------------------------------------------
__global__ void ema_scan_kernel(const float* __restrict__ x,
                                const float* __restrict__ x_mix_last,
                                const float* __restrict__ factor_param,
                                bf16_t* __restrict__ A) {
    const int idx = blockIdx.x * blockDim.x + threadIdx.x;   // 0 .. B*D-1
    const int d = idx % DIM;
    const int b = idx / DIM;

    const float p = factor_param[d];
    const float f = 1.0f / (1.0f + __expf(-p));              // sigmoid
    const float one_mf = 1.0f - f;

    float conv  = 0.0f;
    float carry = x_mix_last[b * DIM + d] * f;               // x_mix_last * f^(t+1)

    const float* xp = x + (size_t)b * DIM + d;
    bf16_t*      ap = A + (size_t)b * DIM + d;               // A[(t*B+b)*D + d]
    const size_t stride = (size_t)BATCH * DIM;

    for (int t = 0; t < L_SEQ; ++t) {
        conv = __builtin_fmaf(f, conv, xp[(size_t)t * stride]);
        const float xm = __builtin_fmaf(one_mf, conv, carry);
        ap[(size_t)t * stride] = (bf16_t)xm;
        carry *= f;
    }
}

// ---------------------------------------------------------------------------
// Kernel 2: W (fp32, [N,K] row-major) -> bf16 (B-operand source; rows of W are
// columns of the K x N B matrix, so row-major W matches the WMMA lane layout).
// ---------------------------------------------------------------------------
__global__ void convert_w_kernel(const float* __restrict__ W,
                                 bf16_t* __restrict__ Wb) {
    const size_t i = ((size_t)blockIdx.x * blockDim.x + threadIdx.x) * 4;
    const v4f v = *(const v4f*)(W + i);
    v4bf o;
    o[0] = (bf16_t)v[0]; o[1] = (bf16_t)v[1];
    o[2] = (bf16_t)v[2]; o[3] = (bf16_t)v[3];
    *(v4bf*)(Wb + i) = o;
}

// ---------------------------------------------------------------------------
// Kernel 3: bf16 WMMA GEMM. out[M,N] = A[M,K] * Wb[N,K]^T, fp32 accumulate.
// 128x128 block tile, 8 waves (2 in M x 4 in N), 64x32 per wave, K-step 32,
// double-buffered LDS (padded stride 40 -> conflict-free b128 DS traffic).
// Tile fill uses async DMA global->LDS when available (ASYNCcnt), else
// register-staged global_load_b128 + ds_store_b128.
// ---------------------------------------------------------------------------
#define BK      32
#define STRD    40                  // padded LDS row stride (elements)
#define TILE_M  128
#define TILE_N  128
#define NKT     (DIM / BK)          // 64 K-steps

__launch_bounds__(256, 2)
__global__ void ema_gemm_kernel(const bf16_t* __restrict__ A,
                                const bf16_t* __restrict__ Bw,
                                float* __restrict__ out) {
    __shared__ bf16_t sA[2][TILE_M * STRD];
    __shared__ bf16_t sB[2][TILE_N * STRD];

    const int tid  = threadIdx.x;
    const int lane = tid & 31;
    const int wave = tid >> 5;          // 0..7
    const int wm   = wave & 1;          // wave tile position in M (0..1)
    const int wn   = wave >> 1;         // wave tile position in N (0..3)
    const int half = lane >> 4;         // half-wave select (K phase per ISA layout)
    const int r    = lane & 15;         // row within 16x16 fragment

    const int m0 = blockIdx.y * TILE_M;
    const int n0 = blockIdx.x * TILE_N;

    // Tile fill: 128 rows x 32 bf16 = 512 x 16B chunks per matrix; 2 per thread.
    const int lr = tid >> 2;            // 0..63
    const int lc = (tid & 3) * 8;       // 0,8,16,24 (elements)

    auto load_tiles = [&](int kt, int buf) {
        const size_t kb = (size_t)kt * BK + lc;
        const bf16_t* ga = A  + (size_t)(m0 + lr) * DIM + kb;
        const bf16_t* gb = Bw + (size_t)(n0 + lr) * DIM + kb;
#if USE_ASYNC_LDS
        __builtin_amdgcn_global_load_async_to_lds_b128(
            (gas_ptr)ga,                      (las_ptr)&sA[buf][lr * STRD + lc],        0, 0);
        __builtin_amdgcn_global_load_async_to_lds_b128(
            (gas_ptr)(ga + (size_t)64 * DIM), (las_ptr)&sA[buf][(lr + 64) * STRD + lc], 0, 0);
        __builtin_amdgcn_global_load_async_to_lds_b128(
            (gas_ptr)gb,                      (las_ptr)&sB[buf][lr * STRD + lc],        0, 0);
        __builtin_amdgcn_global_load_async_to_lds_b128(
            (gas_ptr)(gb + (size_t)64 * DIM), (las_ptr)&sB[buf][(lr + 64) * STRD + lc], 0, 0);
#else
        const v8bf a0 = *(const v8bf*)ga;
        const v8bf a1 = *(const v8bf*)(ga + (size_t)64 * DIM);
        const v8bf b0 = *(const v8bf*)gb;
        const v8bf b1 = *(const v8bf*)(gb + (size_t)64 * DIM);
        *(v8bf*)&sA[buf][lr * STRD + lc]        = a0;
        *(v8bf*)&sA[buf][(lr + 64) * STRD + lc] = a1;
        *(v8bf*)&sB[buf][lr * STRD + lc]        = b0;
        *(v8bf*)&sB[buf][(lr + 64) * STRD + lc] = b1;
#endif
    };

    auto wait_fill = [&]() {
#if USE_ASYNC_LDS
#if __has_builtin(__builtin_amdgcn_s_wait_asynccnt)
        __builtin_amdgcn_s_wait_asynccnt(0);
#else
        asm volatile("s_wait_asynccnt 0" ::: "memory");
#endif
#endif
    };

    v8f acc[4][2];
#pragma unroll
    for (int i = 0; i < 4; ++i)
#pragma unroll
        for (int j = 0; j < 2; ++j)
            acc[i][j] = (v8f)0.0f;

    load_tiles(0, 0);
    wait_fill();
    __syncthreads();

    union FragU { v16bf v; v8bf h[2]; };

    for (int kt = 0; kt < NKT; ++kt) {
        const int cur = kt & 1;

        if (kt + 1 < NKT) {
            load_tiles(kt + 1, cur ^ 1);  // DMA fills next buffer during compute
            if (kt + 4 < NKT) {           // pull A ahead into L2 (global_prefetch_b8)
                __builtin_prefetch(A + (size_t)(m0 + lr) * DIM +
                                   (size_t)(kt + 4) * BK + lc, 0, 1);
            }
        }

        // ISA 16-bit A/B fragment layout: lane<16 holds K[0..7],K[16..23];
        // lane>=16 holds K[8..15],K[24..31] -> two ds_load_b128 per fragment.
        FragU bfr[2];
#pragma unroll
        for (int j = 0; j < 2; ++j) {
            const bf16_t* p = &sB[cur][(wn * 32 + j * 16 + r) * STRD];
            bfr[j].h[0] = *(const v8bf*)(p + half * 8);
            bfr[j].h[1] = *(const v8bf*)(p + 16 + half * 8);
        }
        FragU afr[4];
#pragma unroll
        for (int i = 0; i < 4; ++i) {
            const bf16_t* p = &sA[cur][(wm * 64 + i * 16 + r) * STRD];
            afr[i].h[0] = *(const v8bf*)(p + half * 8);
            afr[i].h[1] = *(const v8bf*)(p + 16 + half * 8);
        }

#pragma unroll
        for (int i = 0; i < 4; ++i)
#pragma unroll
            for (int j = 0; j < 2; ++j)
                acc[i][j] = __builtin_amdgcn_wmma_f32_16x16x32_bf16(
                    false, afr[i].v, false, bfr[j].v,
                    (short)0, acc[i][j], false, false);

        wait_fill();       // next buffer's DMA done (overlapped with WMMAs above)
        __syncthreads();
    }

    // C/D layout: lanes 0-15 hold M = vgpr, lanes 16-31 hold M = 8 + vgpr; N = r.
#pragma unroll
    for (int i = 0; i < 4; ++i) {
#pragma unroll
        for (int j = 0; j < 2; ++j) {
            const int row0 = m0 + wm * 64 + i * 16 + half * 8;
            const int col  = n0 + wn * 32 + j * 16 + r;
            float* op = out + (size_t)row0 * DIM + col;
#pragma unroll
            for (int v = 0; v < 8; ++v)
                op[(size_t)v * DIM] = acc[i][j][v];
        }
    }
}

// ---------------------------------------------------------------------------
extern "C" void kernel_launch(void* const* d_in, const int* in_sizes, int n_in,
                              void* d_out, int out_size, void* d_ws, size_t ws_size,
                              hipStream_t stream) {
    const float* x           = (const float*)d_in[0];   // [L,B,D]
    const float* x_mix_last  = (const float*)d_in[1];   // [B,D]
    const float* factor_par  = (const float*)d_in[2];   // [D]
    const float* W           = (const float*)d_in[3];   // [D,D]
    float*       out         = (float*)d_out;           // [L,B,D]

    bf16_t* Abf = (bf16_t*)d_ws;                                            // 64 MB
    bf16_t* Wbf = (bf16_t*)((char*)d_ws + (size_t)MROWS * DIM * sizeof(bf16_t)); // +8 MB

    // 1) EMA scan -> bf16 A matrix in workspace
    ema_scan_kernel<<<(BATCH * DIM) / 256, 256, 0, stream>>>(
        x, x_mix_last, factor_par, Abf);

    // 2) W -> bf16 (independent of scan; same stream keeps ordering for GEMM)
    convert_w_kernel<<<(DIM * DIM / 4) / 256, 256, 0, stream>>>(W, Wbf);

    // 3) WMMA GEMM: out = A * W^T
    dim3 grid(DIM / TILE_N, MROWS / TILE_M);   // (16, 128)
    ema_gemm_kernel<<<grid, 256, 0, stream>>>(Abf, Wbf, out);
}
